// FunctionEstimator_13529146983179
// MI455X (gfx1250) — matrix operand order repaired and checked by reading, not compiled, not run on hardware
//
#include <hip/hip_runtime.h>
#include <hip/hip_bf16.h>

// ---------------------------------------------------------------------------
// Types for CDNA5 WMMA (wave32, 16x16x32 f16 -> f32)
// ---------------------------------------------------------------------------
typedef __attribute__((ext_vector_type(8)))  _Float16 v8h;
typedef __attribute__((ext_vector_type(16))) _Float16 v16h;
typedef __attribute__((ext_vector_type(8)))  float    v8f;

#define ACT_NONE 0
#define ACT_RELU 1
#define ACT_TANH 2

// ---------------------------------------------------------------------------
// CDNA5 async global->LDS copy (VGLOBAL GLOBAL_LOAD_ASYNC_TO_LDS_B128,
// tracked by ASYNCcnt; see CDNA5 ISA).
// lds_off = wave-relative LDS byte address (low 32 bits of generic pointer),
// g       = per-lane global source address (16 bytes copied per lane).
// ---------------------------------------------------------------------------
__device__ __forceinline__ void async_copy_b128(uint32_t lds_off, const void* g) {
  asm volatile("global_load_async_to_lds_b128 %0, %1, off"
               :: "v"(lds_off), "v"(g) : "memory");
}
__device__ __forceinline__ void wait_async0() {
  asm volatile("s_wait_asynccnt 0" ::: "memory");
}
__device__ __forceinline__ uint32_t lds_addr(const void* p) {
  return (uint32_t)(uintptr_t)p;   // generic LDS pointer: offset in low 32 bits
}

// ---------------------------------------------------------------------------
// Fragment loaders matching CDNA5 ISA VGPR layouts (CDNA5 ISA)
//
// A (16x32 f16): lane l holds row m = l&15.
//   elems 0..7  = K = kb .. kb+7,   kb = (l<16)?0:8
//   elems 8..15 = K = 16+kb .. 16+kb+7
// B (32x16 f16) from *transposed* storage bT[n][k] (i.e. row-major N x K):
//   lane l holds col n = l&15; elems 0..7 = K = kb..kb+7, elems 8..15 = K = 16+kb..16+kb+7, kb = (l<16)?0:8
// C/D (16x16 f32): lane l holds col n = l&15; vgpr i holds row m = i + ((l<16)?0:8)
// ---------------------------------------------------------------------------
__device__ __forceinline__ v16h load_frag_a(const _Float16* p, int stride, int lane) {
  int m  = lane & 15;
  int kb = (lane >> 4) << 3;              // 0 or 8
  const _Float16* r = p + m * stride;
  v16h a;
  ((v8h*)&a)[0] = *(const v8h*)(r + kb);
  ((v8h*)&a)[1] = *(const v8h*)(r + 16 + kb);
  return a;
}

__device__ __forceinline__ v16h load_frag_bT(const _Float16* p, int stride, int lane) {
  int n  = lane & 15;
  int kb = (lane >> 4) << 3;              // 0 or 8
  const _Float16* r = p + n * stride + kb;
  v16h b;
  ((v8h*)&b)[0] = *(const v8h*)(r);
  ((v8h*)&b)[1] = *(const v8h*)(r + 16);
  return b;
}

__device__ __forceinline__ v8f wmma_f16(const v16h& a, const v16h& b, const v8f& c) {
  return __builtin_amdgcn_wmma_f32_16x16x32_f16(false, a, false, b, (short)0, c, false, false);
}

// ---------------------------------------------------------------------------
// Generic WMMA GEMM:  C(MxN) = A(MxK, f16) * W(NxK, f16)^T + bias, act
// Block tile 128x64, 8 waves, wave tile 32x32 (4 accumulators), K-step 32.
// Tiles staged with GLOBAL_LOAD_ASYNC_TO_LDS_B128 (no VGPR round trip).
// M % 128 == 0, N % 64 == 0, K % 32 == 0 by construction for all call sites.
// ---------------------------------------------------------------------------
template<int ACT, bool OUT32, bool OUT16>
__global__ __launch_bounds__(256)
void gemm_f16_kernel(const _Float16* __restrict__ A,
                     const _Float16* __restrict__ W,
                     const float*    __restrict__ bias,
                     float*          __restrict__ C32,
                     _Float16*       __restrict__ C16,
                     int M, int N, int K)
{
  const int SA = 40, SB = 40;                 // padded LDS strides (f16 elems)
  __shared__ __align__(16) _Float16 lds_a[128 * 40];
  __shared__ __align__(16) _Float16 lds_b[64 * 40];   // stores W rows: [n][k]

  int tid  = threadIdx.x;
  int lane = tid & 31;
  int w    = tid >> 5;
  int bm   = blockIdx.y * 128;
  int bn   = blockIdx.x * 64;
  int wm   = (w >> 1) * 32;                   // 0,32,64,96
  int wn   = (w & 1) * 32;                    // 0,32

  // per-thread staging coordinates (fixed across K loop)
  int ea = tid * 16;
  int ra = ea >> 5, ca = ea & 31;             // A: two 16B chunks, ca in {0,16}
  int eb = tid * 8;
  int rb = eb >> 5, cb = eb & 31;             // B: one 16B chunk
  uint32_t la0 = lds_addr(&lds_a[ra * SA + ca]);
  uint32_t lb0 = lds_addr(&lds_b[rb * SB + cb]);
  const _Float16* ga = A + (size_t)(bm + ra) * K + ca;
  const _Float16* gb = W + (size_t)(bn + rb) * K + cb;

  v8f acc[2][2] = {};

  for (int k0 = 0; k0 < K; k0 += 32) {
    // --- async stage A tile 128x32 and W tile 64x32 directly into LDS ---
    async_copy_b128(la0,      ga + k0);
    async_copy_b128(la0 + 16, ga + k0 + 8);
    async_copy_b128(lb0,      gb + k0);
    wait_async0();
    __syncthreads();

    v16h af0 = load_frag_a (&lds_a[(wm)      * SA], SA, lane);
    v16h af1 = load_frag_a (&lds_a[(wm + 16) * SA], SA, lane);
    v16h bf0 = load_frag_bT(&lds_b[(wn)      * SB], SB, lane);
    v16h bf1 = load_frag_bT(&lds_b[(wn + 16) * SB], SB, lane);

    acc[0][0] = wmma_f16(af0, bf0, acc[0][0]);
    acc[0][1] = wmma_f16(af0, bf1, acc[0][1]);
    acc[1][0] = wmma_f16(af1, bf0, acc[1][0]);
    acc[1][1] = wmma_f16(af1, bf1, acc[1][1]);

    __syncthreads();
  }

  // --- epilogue: bias + activation + store ---
  int nc = lane & 15;
  int mb = (lane >> 4) << 3;
  for (int mi = 0; mi < 2; ++mi)
    for (int ni = 0; ni < 2; ++ni) {
      int gn = bn + wn + ni * 16 + nc;
      float bv = bias ? bias[gn] : 0.f;
      for (int i = 0; i < 8; ++i) {
        int gm = bm + wm + mi * 16 + mb + i;
        float v = acc[mi][ni][i] + bv;
        if (ACT == ACT_RELU) v = v > 0.f ? v : 0.f;
        if (ACT == ACT_TANH) v = tanhf(v);
        size_t o = (size_t)gm * N + gn;
        if (OUT32) C32[o] = v;
        if (OUT16) C16[o] = (_Float16)v;
      }
    }
}

// ---------------------------------------------------------------------------
// Fused attention: one block per (n, h).  S=64, Dh=64, softmax over t in LDS.
// qkvh rows are (s*1024+n) x 1536 with [q|k|v] each 512, head h at h*64.
// Q/K tiles staged with async global->LDS; V needs a transpose so it is
// staged element-wise.
// ---------------------------------------------------------------------------
__global__ __launch_bounds__(128)
void attention_kernel(const _Float16* __restrict__ qkvh,
                      _Float16*       __restrict__ Oh,
                      float scale)
{
  __shared__ __align__(16) _Float16 lds_q [64 * 72];
  __shared__ __align__(16) _Float16 lds_k [64 * 72];   // K as [t][d] (B^T form)
  __shared__ __align__(16) _Float16 lds_vT[64 * 72];   // V as [d][t] (B^T form)
  __shared__ __align__(16) _Float16 lds_p [64 * 72];   // probs f16 [s][t]
  __shared__ float lds_s[64 * 65];                     // scores f32

  int n = blockIdx.x, h = blockIdx.y;
  int tid = threadIdx.x, lane = tid & 31, w = tid >> 5;

  // async-stage Q and K tiles (row s contiguous over d, 16B chunks)
  for (int e = tid; e < 64 * 8; e += 128) {
    int s = e >> 3, c = (e & 7) * 8;
    size_t base = ((size_t)(s * 1024 + n)) * 1536 + h * 64 + c;
    async_copy_b128(lds_addr(&lds_q[s * 72 + c]), qkvh + base);
    async_copy_b128(lds_addr(&lds_k[s * 72 + c]), qkvh + base + 512);
  }
  // load V transposed (element-wise)
  for (int e = tid; e < 64 * 64; e += 128) {
    int t = e >> 6, d = e & 63;
    lds_vT[d * 72 + t] = qkvh[((size_t)(t * 1024 + n)) * 1536 + 1024 + h * 64 + d];
  }
  wait_async0();
  __syncthreads();

  // scores = Q x K^T : wave w owns rows [w*16, w*16+16)
  {
    v8f acc[4] = {};
    for (int k0 = 0; k0 < 64; k0 += 32) {
      v16h a = load_frag_a(&lds_q[(w * 16) * 72 + k0], 72, lane);
      for (int ni = 0; ni < 4; ++ni) {
        v16h b = load_frag_bT(&lds_k[(ni * 16) * 72 + k0], 72, lane);
        acc[ni] = wmma_f16(a, b, acc[ni]);
      }
    }
    int nc = lane & 15, mb = (lane >> 4) << 3;
    for (int ni = 0; ni < 4; ++ni)
      for (int i = 0; i < 8; ++i)
        lds_s[(w * 16 + mb + i) * 65 + ni * 16 + nc] = acc[ni][i];
  }
  __syncthreads();

  // softmax per row (threads 0..63 each own a row of 64)
  if (tid < 64) {
    float mx = -1e30f;
    for (int t = 0; t < 64; ++t) {
      float v = lds_s[tid * 65 + t] * scale;
      lds_s[tid * 65 + t] = v;
      mx = fmaxf(mx, v);
    }
    float sum = 0.f;
    for (int t = 0; t < 64; ++t) {
      float e = __expf(lds_s[tid * 65 + t] - mx);
      lds_s[tid * 65 + t] = e;
      sum += e;
    }
    float inv = 1.f / sum;
    for (int t = 0; t < 64; ++t)
      lds_p[tid * 72 + t] = (_Float16)(lds_s[tid * 65 + t] * inv);
  }
  __syncthreads();

  // O = P x V
  {
    v8f acc[4] = {};
    for (int k0 = 0; k0 < 64; k0 += 32) {
      v16h a = load_frag_a(&lds_p[(w * 16) * 72 + k0], 72, lane);
      for (int ni = 0; ni < 4; ++ni) {
        v16h b = load_frag_bT(&lds_vT[(ni * 16) * 72 + k0], 72, lane);
        acc[ni] = wmma_f16(a, b, acc[ni]);
      }
    }
    int nc = lane & 15, mb = (lane >> 4) << 3;
    for (int ni = 0; ni < 4; ++ni)
      for (int i = 0; i < 8; ++i) {
        int s = w * 16 + mb + i, d = ni * 16 + nc;
        Oh[((size_t)(s * 1024 + n)) * 512 + h * 64 + d] = (_Float16)acc[ni][i];
      }
  }
}

// ---------------------------------------------------------------------------
// Reductions / elementwise helpers (wave32)
// ---------------------------------------------------------------------------
__device__ __forceinline__ float block_sum_256(float v, float* red, int tid) {
  for (int o = 16; o > 0; o >>= 1) v += __shfl_down(v, o, 32);
  if ((tid & 31) == 0) red[tid >> 5] = v;
  __syncthreads();
  if (tid < 32) {
    float r = (tid < 8) ? red[tid] : 0.f;
    for (int o = 4; o > 0; o >>= 1) r += __shfl_down(r, o, 32);
    if (tid == 0) red[15] = r;
  }
  __syncthreads();
  float out = red[15];
  __syncthreads();
  return out;
}

__global__ __launch_bounds__(256)
void ln_kernel(const float* __restrict__ x, const float* __restrict__ delta,
               const float* __restrict__ g, const float* __restrict__ b,
               float* __restrict__ xo, _Float16* __restrict__ xo16)
{
  __shared__ float red[16];
  size_t row = blockIdx.x;
  int tid = threadIdx.x;
  const float* xr = x + row * 512;
  const float* dr = delta + row * 512;
  float v0 = xr[tid] + dr[tid];
  float v1 = xr[tid + 256] + dr[tid + 256];
  float mu = block_sum_256(v0 + v1, red, tid) * (1.f / 512.f);
  float d0 = v0 - mu, d1 = v1 - mu;
  float var = block_sum_256(d0 * d0 + d1 * d1, red, tid) * (1.f / 512.f);
  float rs = rsqrtf(var + 1e-5f);
  float o0 = d0 * rs * g[tid]       + b[tid];
  float o1 = d1 * rs * g[tid + 256] + b[tid + 256];
  xo[row * 512 + tid]         = o0;
  xo[row * 512 + tid + 256]   = o1;
  xo16[row * 512 + tid]       = (_Float16)o0;
  xo16[row * 512 + tid + 256] = (_Float16)o1;
}

__global__ __launch_bounds__(256)
void minmax_kernel(const float* __restrict__ values, float* __restrict__ vmm) {
  __shared__ float rmin[256], rmax[256];
  int b = blockIdx.x, tid = threadIdx.x;
  float mn = 1e30f, mx = -1e30f;
  for (int i = tid; i < 1024; i += 256) {
    float v = values[b * 1024 + i];
    mn = fminf(mn, v); mx = fmaxf(mx, v);
  }
  rmin[tid] = mn; rmax[tid] = mx;
  __syncthreads();
  for (int o = 128; o > 0; o >>= 1) {
    if (tid < o) {
      rmin[tid] = fminf(rmin[tid], rmin[tid + o]);
      rmax[tid] = fmaxf(rmax[tid], rmax[tid + o]);
    }
    __syncthreads();
  }
  if (tid == 0) { vmm[b] = rmin[0]; vmm[64 + b] = rmax[0]; }
}

__global__ __launch_bounds__(256)
void embed_kernel(const float* __restrict__ ii, const float* __restrict__ vals,
                  const float* __restrict__ vmm,
                  const float* __restrict__ w_idx, const float* __restrict__ b_idx,
                  const float* __restrict__ w_val, const float* __restrict__ b_val,
                  float* __restrict__ X, _Float16* __restrict__ XH)
{
  size_t idx = (size_t)blockIdx.x * 256 + threadIdx.x;   // 65536*512
  int c = (int)(idx & 511);
  size_t row = idx >> 9;
  int b = (int)(row >> 10), l = (int)(row & 1023);
  float v;
  if (c < 256) {
    float t = ii[b * 1024 + l] * (1.f / 1024.f);
    v = w_idx[c] * t + b_idx[c];
  } else {
    float vmin = vmm[b], vmax = vmm[64 + b];
    float t = (vals[b * 1024 + l] - vmin) / (vmax - vmin);
    v = w_val[c - 256] * t + b_val[c - 256];
  }
  X[idx] = v;
  XH[idx] = (_Float16)v;
}

__global__ __launch_bounds__(256)
void convert_f32_f16(const float* __restrict__ src, _Float16* __restrict__ dst, size_t n) {
  size_t i = (size_t)blockIdx.x * 256 + threadIdx.x;
  if (i < n) dst[i] = (_Float16)src[i];
}

__global__ __launch_bounds__(256)
void oiproj_kernel(const float* __restrict__ oi, const float* __restrict__ w2,
                   const float* __restrict__ b2, _Float16* __restrict__ O) {
  int idx = blockIdx.x * 256 + threadIdx.x;     // 1024*256
  int c = idx & 255, j = idx >> 8;
  float t = oi[j] * (1.f / 1024.f);
  O[idx] = (_Float16)(t * w2[c] + b2[c]);
}

__global__ __launch_bounds__(256)
void decin_kernel(const _Float16* __restrict__ XH, const _Float16* __restrict__ OIPH,
                  _Float16* __restrict__ D) {
  size_t idx = (size_t)blockIdx.x * 256 + threadIdx.x;   // 65536*768
  int c = (int)(idx % 768);
  size_t r = idx / 768;
  int s = (int)(r >> 10), j = (int)(r & 1023);
  D[idx] = (c < 512) ? XH[(size_t)s * 1024 * 512 + c] : OIPH[j * 256 + (c - 512)];
}

__global__ __launch_bounds__(256)
void head_kernel(const _Float16* __restrict__ H, const float* __restrict__ hw,
                 const float* __restrict__ hb, float* __restrict__ out) {
  __shared__ float red[16];
  size_t row = blockIdx.x;
  int tid = threadIdx.x;
  float s = 0.f;
  for (int i = tid; i < 1536; i += 256) s += (float)H[row * 1536 + i] * hw[i];
  for (int o = 16; o > 0; o >>= 1) s += __shfl_down(s, o, 32);
  if ((tid & 31) == 0) red[tid >> 5] = s;
  __syncthreads();
  if (tid < 32) {
    float r = (tid < 8) ? red[tid] : 0.f;
    for (int o = 4; o > 0; o >>= 1) r += __shfl_down(r, o, 32);
    if (tid == 0) out[row] = r + hb[0];
  }
}

__global__ void tail_zero(float* out) { if (threadIdx.x == 0) out[65536] = 0.f; }

// ---------------------------------------------------------------------------
// Host orchestration
// ---------------------------------------------------------------------------
extern "C" void kernel_launch(void* const* d_in, const int* in_sizes, int n_in,
                              void* d_out, int out_size, void* d_ws, size_t ws_size,
                              hipStream_t stream) {
  const float* input_indices = (const float*)d_in[0];
  const float* values        = (const float*)d_in[1];
  const float* output_idx    = (const float*)d_in[2];
  const float* w_idx  = (const float*)d_in[3];
  const float* b_idx  = (const float*)d_in[4];
  const float* w_val  = (const float*)d_in[5];
  const float* b_val  = (const float*)d_in[6];
  const float* w_idx2 = (const float*)d_in[7];
  const float* b_idx2 = (const float*)d_in[8];
  const float* qkv_w  = (const float*)d_in[9];
  const float* qkv_b  = (const float*)d_in[10];
  const float* out_w  = (const float*)d_in[11];
  const float* out_b  = (const float*)d_in[12];
  const float* ln1_g  = (const float*)d_in[13];
  const float* ln1_b  = (const float*)d_in[14];
  const float* ln2_g  = (const float*)d_in[15];
  const float* ln2_b  = (const float*)d_in[16];
  const float* ff1_w  = (const float*)d_in[17];
  const float* ff1_b  = (const float*)d_in[18];
  const float* ff2_w  = (const float*)d_in[19];
  const float* ff2_b  = (const float*)d_in[20];
  const float* dec1_w = (const float*)d_in[21];
  const float* dec1_b = (const float*)d_in[22];
  const float* dec2_w = (const float*)d_in[23];
  const float* dec2_b = (const float*)d_in[24];
  const float* dec3_w = (const float*)d_in[25];
  const float* dec3_b = (const float*)d_in[26];
  const float* head_w = (const float*)d_in[27];
  const float* head_b = (const float*)d_in[28];

  const size_t M = 65536;
  char* ws = (char*)d_ws;
  size_t off = 0;
  auto carve = [&](size_t bytes) { char* p = ws + off; off += (bytes + 255) & ~(size_t)255; return p; };

  float*    X    = (float*)   carve(M * 512 * 4);     // activations f32
  _Float16* XH   = (_Float16*)carve(M * 512 * 2);     // activations f16
  _Float16* BUF1 = (_Float16*)carve(M * 3072 * 2);    // qkvh / dec2-out
  _Float16* BUF2 = (_Float16*)carve(M * 2048 * 2);    // attn-out / ff1-out / dec_in
  float*    BUF3 = (float*)   carve(M * 512 * 4);     // proj / ff2 out f32
  _Float16* BUF4 = (_Float16*)carve(M * 1536 * 2);    // dec1-out / dec3-out
  _Float16* WTH  = (_Float16*)carve((size_t)3072 * 1536 * 2);  // converted weight
  float*    VMM  = (float*)   carve(128 * 4);
  _Float16* OIPH = (_Float16*)carve(1024 * 256 * 2);

  auto cvt = [&](const float* src, _Float16* dst, size_t n) {
    convert_f32_f16<<<(unsigned)((n + 255) / 256), 256, 0, stream>>>(src, dst, n);
  };

  // ---- embedding ----
  minmax_kernel<<<64, 256, 0, stream>>>(values, VMM);
  embed_kernel<<<(unsigned)(M * 512 / 256), 256, 0, stream>>>(
      input_indices, values, VMM, w_idx, b_idx, w_val, b_val, X, XH);

  // ---- transformer layers ----
  const float scale = 0.125f;  // 1/sqrt(64)
  for (int l = 0; l < 6; ++l) {
    // QKV: (M,512) x (1536,512)^T -> f16
    cvt(qkv_w + (size_t)l * 1536 * 512, WTH, (size_t)1536 * 512);
    gemm_f16_kernel<ACT_NONE, false, true><<<dim3(1536 / 64, M / 128), 256, 0, stream>>>(
        XH, WTH, qkv_b + l * 1536, nullptr, BUF1, (int)M, 1536, 512);

    // fused attention over S=64 axis
    attention_kernel<<<dim3(1024, 8), 128, 0, stream>>>(BUF1, BUF2, scale);

    // out-proj: (M,512) x (512,512)^T -> f32
    cvt(out_w + (size_t)l * 512 * 512, WTH, (size_t)512 * 512);
    gemm_f16_kernel<ACT_NONE, true, false><<<dim3(512 / 64, M / 128), 256, 0, stream>>>(
        BUF2, WTH, out_b + l * 512, BUF3, nullptr, (int)M, 512, 512);

    ln_kernel<<<(unsigned)M, 256, 0, stream>>>(X, BUF3, ln1_g + l * 512, ln1_b + l * 512, X, XH);

    // FF1 + ReLU -> f16
    cvt(ff1_w + (size_t)l * 2048 * 512, WTH, (size_t)2048 * 512);
    gemm_f16_kernel<ACT_RELU, false, true><<<dim3(2048 / 64, M / 128), 256, 0, stream>>>(
        XH, WTH, ff1_b + l * 2048, nullptr, BUF2, (int)M, 2048, 512);

    // FF2 -> f32
    cvt(ff2_w + (size_t)l * 512 * 2048, WTH, (size_t)512 * 2048);
    gemm_f16_kernel<ACT_NONE, true, false><<<dim3(512 / 64, M / 128), 256, 0, stream>>>(
        BUF2, WTH, ff2_b + l * 512, BUF3, nullptr, (int)M, 512, 2048);

    ln_kernel<<<(unsigned)M, 256, 0, stream>>>(X, BUF3, ln2_g + l * 512, ln2_b + l * 512, X, XH);
  }

  // ---- decoder ----
  oiproj_kernel<<<1024 * 256 / 256, 256, 0, stream>>>(output_idx, w_idx2, b_idx2, OIPH);
  decin_kernel<<<(unsigned)(M * 768 / 256), 256, 0, stream>>>(XH, OIPH, BUF2);

  cvt(dec1_w, WTH, (size_t)1536 * 768);
  gemm_f16_kernel<ACT_TANH, false, true><<<dim3(1536 / 64, M / 128), 256, 0, stream>>>(
      BUF2, WTH, dec1_b, nullptr, BUF4, (int)M, 1536, 768);

  cvt(dec2_w, WTH, (size_t)3072 * 1536);
  gemm_f16_kernel<ACT_TANH, false, true><<<dim3(3072 / 64, M / 128), 256, 0, stream>>>(
      BUF4, WTH, dec2_b, nullptr, BUF1, (int)M, 3072, 1536);

  cvt(dec3_w, WTH, (size_t)1536 * 3072);
  gemm_f16_kernel<ACT_TANH, false, true><<<dim3(1536 / 64, M / 128), 256, 0, stream>>>(
      BUF1, WTH, dec3_b, nullptr, BUF4, (int)M, 1536, 3072);

  head_kernel<<<(unsigned)M, 256, 0, stream>>>(BUF4, head_w, head_b, (float*)d_out);
  tail_zero<<<1, 32, 0, stream>>>((float*)d_out);
}
